// BiAttention_6665789244009
// MI455X (gfx1250) — matrix operand order repaired and checked by reading, not compiled, not run on hardware
//
#include <hip/hip_runtime.h>
#include <hip/hip_bf16.h>
#include <stdint.h>

// ---------------- problem constants ----------------
#define BN   8      // batch
#define LD   2048   // input rows
#define LM   512    // memory rows
#define DD   1024   // feature dim
#define HD   1024   // hidden dim
#define NEGC 1e30f

typedef __attribute__((ext_vector_type(16))) __bf16 v16bf;
typedef __attribute__((ext_vector_type(8)))  __bf16 v8bf;
typedef __attribute__((ext_vector_type(8)))  float  v8f;
typedef __attribute__((ext_vector_type(4)))  unsigned int u32x4;
typedef __attribute__((ext_vector_type(8)))  int i32x8;
typedef __attribute__((ext_vector_type(4)))  int i32x4;

// ---------------- WMMA fragment helpers (wave32, 16x16x32 bf16) ----------------
__device__ __forceinline__ v16bf frag_combine(const __bf16* p0, const __bf16* p1) {
    v8bf lo = *(const v8bf*)p0;   // 16B aligned
    v8bf hi = *(const v8bf*)p1;
    v16bf r;
#pragma unroll
    for (int i = 0; i < 8; ++i) { r[i] = lo[i]; r[i + 8] = hi[i]; }
    return r;
}

// A matrix 16x32 (MxK), row-major with row stride ldk (elements).
__device__ __forceinline__ v16bf load_a_frag(const __bf16* base, int ldk, int lane) {
    const int row  = lane & 15;
    const int half = lane >> 4;
    const __bf16* r = base + (long)row * ldk;
    return frag_combine(r + half * 8, r + 16 + half * 8);
}

// B matrix 32x16 (KxN) supplied as B^T row-major [n][k] with row stride ldk.
__device__ __forceinline__ v16bf load_b_frag(const __bf16* baseT, int ldk, int lane) {
    const int col = lane & 15;
    const int kb  = (lane >> 4) * 8;
    const __bf16* r = baseT + (long)col * ldk + kb;
    return frag_combine(r, r + 16);
}

__device__ __forceinline__ v8f wmma_bf16(v16bf a, v16bf b, v8f c) {
    return __builtin_amdgcn_wmma_f32_16x16x32_bf16(false, a, false, b,
                                                   (short)0, c, false, false);
}

// ---------------- TDM: 2-D tile (elements of 2 bytes) global -> LDS ----------
// D# packing per CDNA5 ISA.
__device__ __forceinline__ void tdm_load_2d(unsigned lds_off, const void* gptr,
                                            unsigned tile_w, unsigned tile_h,
                                            unsigned tensor_w, unsigned tensor_h,
                                            unsigned row_stride) {
    unsigned long long ga = (unsigned long long)(uintptr_t)gptr;
    u32x4 g0 = { 1u,                                   // count = 1 valid descriptor
                 lds_off,                              // lds_addr
                 (unsigned)(ga & 0xffffffffu),         // global_addr[31:0]
                 (unsigned)((ga >> 32) & 0x01ffffffu)  // global_addr[56:32]
                     | (2u << 30) };                   // type = 2 ("image")
    i32x8 g1;
    g1[0] = (int)(1u << 16);                                  // data_size = 2 bytes
    g1[1] = (int)((tensor_w & 0xffffu) << 16);                // tensor_dim0[15:0]
    g1[2] = (int)((tensor_w >> 16) | ((tensor_h & 0xffffu) << 16));
    g1[3] = (int)((tensor_h >> 16) | ((tile_w & 0xffffu) << 16)); // tile_dim0
    g1[4] = (int)(tile_h & 0xffffu);                          // tile_dim1 (dim2 = 0)
    g1[5] = (int)row_stride;                                  // tensor_dim0_stride[31:0]
    g1[6] = 0;
    g1[7] = 0;
    i32x4 z4 = (i32x4)0;
#if __clang_major__ >= 23
    i32x8 z8 = (i32x8)0;
    __builtin_amdgcn_tensor_load_to_lds(g0, g1, z4, z4, z8, 0);
#else
    __builtin_amdgcn_tensor_load_to_lds(g0, g1, z4, z4, 0);
#endif
}

// ---------------- K1: f32 -> bf16 row convert + rank-1 dot ----------------
__global__ __launch_bounds__(256)
void rowconv_kernel(const float* __restrict__ src, const float* __restrict__ w,
                    __bf16* __restrict__ dst, float* __restrict__ dot) {
    __shared__ float red[256];
    const long row = blockIdx.x;
    const float* s = src + row * DD;
    __bf16* d = dst + row * DD;
    float acc = 0.f;
    for (int i = threadIdx.x; i < DD; i += 256) {
        float v = s[i];
        d[i] = (__bf16)v;
        acc += v * w[i];
    }
    red[threadIdx.x] = acc;
    __syncthreads();
    for (int st = 128; st > 0; st >>= 1) {
        if (threadIdx.x < st) red[threadIdx.x] += red[threadIdx.x + st];
        __syncthreads();
    }
    if (threadIdx.x == 0) dot[row] = red[0];
}

// ---------------- K2: weight [d][h] f32 -> [h][d] bf16 ----------------
__global__ __launch_bounds__(256)
void wtrans_kernel(const float* __restrict__ W, __bf16* __restrict__ WT) {
    const int h = blockIdx.x * 256 + threadIdx.x;   // coalesced read
    const int d = blockIdx.y;
    WT[(long)h * DD + d] = (__bf16)W[(long)d * HD + h];
}

// ---------------- K3: GEMM C = A(bf16) * BT(bf16)^T + bias ----------------
// 64x128 WG tile, 8 waves; B tile staged by TDM into double-buffered LDS,
// A fragments register-pipelined from global; B fragments hoisted so the
// 4 WMMAs issue back-to-back behind a single DScnt wait.
__global__ __launch_bounds__(256)
void gemm_bias_kernel(const __bf16* __restrict__ A, const __bf16* __restrict__ BT,
                      const float* __restrict__ bias, float* __restrict__ C,
                      long c_stride, __bf16* __restrict__ Ct, int rowsPerBatchT,
                      int Ksz) {
    __shared__ __bf16 shB[2][128 * 32];   // 2 x 8 KB double buffer
    const int lane = threadIdx.x & 31;
    const int wave = threadIdx.x >> 5;
    const int rt = wave & 3;              // row tile 0..3
    const int cg = wave >> 2;             // col group 0..1
    const long rowBase = (long)blockIdx.x * 64 + rt * 16;
    const int  colWG   = blockIdx.y * 128;
    const int  colBase = colWG + cg * 64;
    const __bf16* Arow  = A + rowBase * Ksz;
    const __bf16* Btile = BT + (long)colWG * Ksz;

    v8f acc[4];
#pragma unroll
    for (int j = 0; j < 4; ++j)
#pragma unroll
        for (int e = 0; e < 8; ++e) acc[j][e] = 0.f;

    const int nsteps = Ksz / 32;
    if (wave == 0)
        tdm_load_2d((unsigned)(uintptr_t)&shB[0][0], Btile,
                    32u, 128u, (unsigned)Ksz, 128u, (unsigned)Ksz);
    v16bf a_cur = load_a_frag(Arow, Ksz, lane);

    for (int s = 0; s < nsteps; ++s) {
        const int k = s * 32;
        if (wave == 0) {
            if (s + 1 < nsteps) {
                tdm_load_2d((unsigned)(uintptr_t)&shB[(s + 1) & 1][0],
                            Btile + k + 32, 32u, 128u,
                            (unsigned)Ksz, 128u, (unsigned)Ksz);
                __builtin_amdgcn_s_wait_tensorcnt((short)1);  // current buf ready
            } else {
                __builtin_amdgcn_s_wait_tensorcnt((short)0);
            }
        }
        __syncthreads();

        // hoist: 8 ds_load_b128 + next-A global loads issued before any WMMA
        const __bf16* sb = &shB[s & 1][0];
        const int kb = (lane >> 4) * 8;
        v16bf b[4];
#pragma unroll
        for (int j = 0; j < 4; ++j) {
            const int cl = cg * 64 + 16 * j + (lane & 15);
            b[j] = frag_combine(sb + cl * 32 + kb, sb + cl * 32 + kb + 16);
        }
        v16bf a_nxt = a_cur;
        if (s + 1 < nsteps) a_nxt = load_a_frag(Arow + k + 32, Ksz, lane);
        __builtin_prefetch(Arow + k + 128, 0, 1);   // global_prefetch_b8

#pragma unroll
        for (int j = 0; j < 4; ++j) acc[j] = wmma_bf16(a_cur, b[j], acc[j]);

        a_cur = a_nxt;
        __syncthreads();   // all reads done before next TDM overwrites this buf
    }

#pragma unroll
    for (int j = 0; j < 4; ++j) {
        const int col = colBase + 16 * j + (lane & 15);
        const float bv = bias ? bias[col] : 0.f;
#pragma unroll
        for (int e = 0; e < 8; ++e) {
            const long row = rowBase + e + (lane >> 4) * 8;
            const float v = acc[j][e] + bv;
            C[row * c_stride + col] = v;
            if (Ct) {   // bf16 transposed copy: Ct[n][h][m]
                const int n = (int)(row / rowsPerBatchT);
                const int m = (int)(row % rowsPerBatchT);
                Ct[((long)n * HD + col) * rowsPerBatchT + m] = (__bf16)v;
            }
        }
    }
}

// ---------------- K4: attention logits + softmax over Lm ----------------
// WG: 16 input rows x 512 memory cols; 8 waves x (16x64); register pipelined.
__global__ __launch_bounds__(256)
void att_kernel(const __bf16* __restrict__ inBF, const __bf16* __restrict__ memBF,
                const float* __restrict__ in_dot, const float* __restrict__ mem_dot,
                const float* __restrict__ mask,
                float* __restrict__ rowmax, __bf16* __restrict__ P) {
    __shared__ float tile[16][LM];    // 32 KB of the 320 KB WGP LDS
    __shared__ float red[16][16];
    const int n   = blockIdx.y;
    const long l0 = (long)blockIdx.x * 16;
    const int lane = threadIdx.x & 31;
    const int wave = threadIdx.x >> 5;

    const __bf16* Arow  = inBF + ((long)n * LD + l0) * DD;
    const __bf16* Mbase = memBF + (long)n * LM * DD;

    v8f acc[4];
#pragma unroll
    for (int j = 0; j < 4; ++j)
#pragma unroll
        for (int e = 0; e < 8; ++e) acc[j][e] = 0.f;

    // software pipeline: hold current fragments, load next before WMMAs
    v16bf a_cur = load_a_frag(Arow, DD, lane);
    v16bf b_cur[4];
#pragma unroll
    for (int j = 0; j < 4; ++j)
        b_cur[j] = load_b_frag(Mbase + (long)(wave * 64 + 16 * j) * DD, DD, lane);

    for (int k = 0; k < DD - 32; k += 32) {
        v16bf a_nxt = load_a_frag(Arow + k + 32, DD, lane);
        v16bf b_nxt[4];
#pragma unroll
        for (int j = 0; j < 4; ++j)
            b_nxt[j] = load_b_frag(Mbase + (long)(wave * 64 + 16 * j) * DD + k + 32,
                                   DD, lane);
#pragma unroll
        for (int j = 0; j < 4; ++j) acc[j] = wmma_bf16(a_cur, b_cur[j], acc[j]);
        a_cur = a_nxt;
#pragma unroll
        for (int j = 0; j < 4; ++j) b_cur[j] = b_nxt[j];
    }
#pragma unroll
    for (int j = 0; j < 4; ++j) acc[j] = wmma_bf16(a_cur, b_cur[j], acc[j]);

    const float inv_scale = 0.03125f;   // 1/sqrt(1024)
#pragma unroll
    for (int j = 0; j < 4; ++j) {
        const int m = wave * 64 + 16 * j + (lane & 15);
        const float md = mem_dot[n * LM + m];
        const float mk = mask[n * LM + m];
#pragma unroll
        for (int e = 0; e < 8; ++e) {
            const int r = e + (lane >> 4) * 8;
            tile[r][m] = acc[j][e] * inv_scale + in_dot[n * LD + l0 + r] + md
                         - NEGC * (1.f - mk);
        }
    }
    __syncthreads();

    // softmax: 16 threads per row, 32 cols each
    const int r = threadIdx.x & 15;
    const int c = threadIdx.x >> 4;
    float mx = -3.4e38f;
    for (int i = 0; i < 32; ++i) mx = fmaxf(mx, tile[r][c * 32 + i]);
    red[r][c] = mx;
    __syncthreads();
    if (threadIdx.x < 16) {
        float m2 = red[threadIdx.x][0];
        for (int i = 1; i < 16; ++i) m2 = fmaxf(m2, red[threadIdx.x][i]);
        red[threadIdx.x][0] = m2;
        rowmax[(long)n * LD + l0 + threadIdx.x] = m2;   // for weight_two
    }
    __syncthreads();
    const float rmx = red[r][0];
    float s = 0.f;
    for (int i = 0; i < 32; ++i) {
        float e = __expf(tile[r][c * 32 + i] - rmx);
        tile[r][c * 32 + i] = e;
        s += e;
    }
    __syncthreads();
    red[r][c] = s;
    __syncthreads();
    if (threadIdx.x < 16) {
        float t = 0.f;
        for (int i = 0; i < 16; ++i) t += red[threadIdx.x][i];
        red[threadIdx.x][0] = t;
    }
    __syncthreads();
    const float inv_s = 1.f / red[r][0];
    __bf16* Pr = P + ((long)n * LD + l0 + r) * LM;
    for (int i = 0; i < 32; ++i)
        Pr[c * 32 + i] = (__bf16)(tile[r][c * 32 + i] * inv_s);
}

// ---------------- K5: weight_two = softmax over Ld of rowmax ----------------
__global__ __launch_bounds__(256)
void w2_kernel(const float* __restrict__ rowmax, float* __restrict__ w2) {
    __shared__ float red[256];
    const int n = blockIdx.x;
    const float* rm = rowmax + (long)n * LD;
    float mx = -3.4e38f;
    for (int i = threadIdx.x; i < LD; i += 256) mx = fmaxf(mx, rm[i]);
    red[threadIdx.x] = mx;
    __syncthreads();
    for (int st = 128; st > 0; st >>= 1) {
        if (threadIdx.x < st) red[threadIdx.x] = fmaxf(red[threadIdx.x], red[threadIdx.x + st]);
        __syncthreads();
    }
    mx = red[0];
    __syncthreads();
    float sum = 0.f;
    for (int i = threadIdx.x; i < LD; i += 256) sum += __expf(rm[i] - mx);
    red[threadIdx.x] = sum;
    __syncthreads();
    for (int st = 128; st > 0; st >>= 1) {
        if (threadIdx.x < st) red[threadIdx.x] += red[threadIdx.x + st];
        __syncthreads();
    }
    const float inv = 1.f / red[0];
    for (int i = threadIdx.x; i < LD; i += 256)
        w2[(long)n * LD + i] = __expf(rm[i] - mx) * inv;
}

// ---------------- K6: output_two[n,h] = sum_l w2[n,l] * inp2[n,l,h] ----------------
__global__ __launch_bounds__(256)
void otwo_kernel(const float* __restrict__ out, const float* __restrict__ w2,
                 float* __restrict__ otwo) {
    const int n = blockIdx.y;
    const int h = blockIdx.x * 256 + threadIdx.x;
    const float* base = out + (long)n * LD * (4 * HD) + h;   // inp2 channel
    const float* w = w2 + (long)n * LD;
    float s = 0.f;
    for (int l = 0; l < LD; ++l) s += w[l] * base[(long)l * (4 * HD)];
    otwo[(long)n * HD + h] = s;
}

// ---------------- K7: output_one = P @ mem2 + fused epilogue ----------------
__global__ __launch_bounds__(256)
void out_gemm_kernel(const __bf16* __restrict__ P, const __bf16* __restrict__ mem2t,
                     const float* __restrict__ otwo, float* __restrict__ out) {
    const int n = blockIdx.z;
    const int lane = threadIdx.x & 31;
    const int wave = threadIdx.x >> 5;
    const int rt = wave & 3;
    const int cg = wave >> 2;
    const long rowBase = (long)blockIdx.x * 64 + rt * 16;      // l
    const int  colBase = blockIdx.y * 128 + cg * 64;           // h
    const __bf16* Arow = P + ((long)n * LD + rowBase) * LM;
    const __bf16* BT   = mem2t + (long)n * HD * LM;

    v8f acc[4];
#pragma unroll
    for (int j = 0; j < 4; ++j)
#pragma unroll
        for (int e = 0; e < 8; ++e) acc[j][e] = 0.f;

    v16bf a_cur = load_a_frag(Arow, LM, lane);
    v16bf b_cur[4];
#pragma unroll
    for (int j = 0; j < 4; ++j)
        b_cur[j] = load_b_frag(BT + (long)(colBase + 16 * j) * LM, LM, lane);

    for (int k = 0; k < LM - 32; k += 32) {
        v16bf a_nxt = load_a_frag(Arow + k + 32, LM, lane);
        v16bf b_nxt[4];
#pragma unroll
        for (int j = 0; j < 4; ++j)
            b_nxt[j] = load_b_frag(BT + (long)(colBase + 16 * j) * LM + k + 32,
                                   LM, lane);
#pragma unroll
        for (int j = 0; j < 4; ++j) acc[j] = wmma_bf16(a_cur, b_cur[j], acc[j]);
        a_cur = a_nxt;
#pragma unroll
        for (int j = 0; j < 4; ++j) b_cur[j] = b_nxt[j];
    }
#pragma unroll
    for (int j = 0; j < 4; ++j) acc[j] = wmma_bf16(a_cur, b_cur[j], acc[j]);

#pragma unroll
    for (int j = 0; j < 4; ++j) {
        const int col = colBase + 16 * j + (lane & 15);
        const float ot = otwo[(long)n * HD + col];
#pragma unroll
        for (int e = 0; e < 8; ++e) {
            const long l = rowBase + e + (lane >> 4) * 8;
            const long ob = ((long)n * LD + l) * (4 * HD);
            const float o1 = acc[j][e];
            const float i2 = out[ob + col];            // inp2 written earlier
            out[ob + HD + col]     = o1;               // output_one
            out[ob + 2 * HD + col] = i2 * o1;          // inp2 * output_one
            out[ob + 3 * HD + col] = ot * o1;          // output_two * output_one
        }
    }
}

// ---------------- host launcher ----------------
extern "C" void kernel_launch(void* const* d_in, const int* in_sizes, int n_in,
                              void* d_out, int out_size, void* d_ws, size_t ws_size,
                              hipStream_t stream) {
    const float* input   = (const float*)d_in[0];
    const float* memory  = (const float*)d_in[1];
    const float* mask    = (const float*)d_in[2];
    const float* in1_w   = (const float*)d_in[3];
    const float* mem1_w  = (const float*)d_in[4];
    const float* in2_w   = (const float*)d_in[5];
    const float* in2_b   = (const float*)d_in[6];
    const float* mem2_w  = (const float*)d_in[7];
    const float* mem2_b  = (const float*)d_in[8];
    float* out = (float*)d_out;

    const long OUT1 = (long)BN * LD * 4 * HD;   // start of mem2 in d_out

    // workspace carve-up (256B aligned)
    char* ws = (char*)d_ws;
    size_t off = 0;
    auto carve = [&](size_t bytes) -> void* {
        void* p = ws + off;
        off = (off + bytes + 255) & ~(size_t)255;
        return p;
    };
    __bf16* inBF    = (__bf16*)carve((size_t)BN * LD * DD * 2);
    __bf16* memBF   = (__bf16*)carve((size_t)BN * LM * DD * 2);
    __bf16* in2wT   = (__bf16*)carve((size_t)HD * DD * 2);
    __bf16* mem2wT  = (__bf16*)carve((size_t)HD * DD * 2);
    __bf16* mem2t   = (__bf16*)carve((size_t)BN * HD * LM * 2);
    __bf16* Pbuf    = (__bf16*)carve((size_t)BN * LD * LM * 2);
    float*  in_dot  = (float*)carve((size_t)BN * LD * 4);
    float*  mem_dot = (float*)carve((size_t)BN * LM * 4);
    float*  rowmax  = (float*)carve((size_t)BN * LD * 4);
    float*  w2v     = (float*)carve((size_t)BN * LD * 4);
    float*  otwo    = (float*)carve((size_t)BN * HD * 4);
    (void)ws_size; (void)n_in; (void)in_sizes; (void)out_size;

    // 1. convert + rank-1 dots
    rowconv_kernel<<<BN * LD, 256, 0, stream>>>(input, in1_w, inBF, in_dot);
    rowconv_kernel<<<BN * LM, 256, 0, stream>>>(memory, mem1_w, memBF, mem_dot);

    // 2. transposed bf16 weights
    wtrans_kernel<<<dim3(HD / 256, DD), 256, 0, stream>>>(in2_w, in2wT);
    wtrans_kernel<<<dim3(HD / 256, DD), 256, 0, stream>>>(mem2_w, mem2wT);

    // 3. inp2 -> out[..,0:HD];  mem2 -> out tail (+ bf16 transposed copy)
    gemm_bias_kernel<<<dim3((BN * LD) / 64, HD / 128), 256, 0, stream>>>(
        inBF, in2wT, in2_b, out, (long)(4 * HD), (__bf16*)nullptr, 1, DD);
    gemm_bias_kernel<<<dim3((BN * LM) / 64, HD / 128), 256, 0, stream>>>(
        memBF, mem2wT, mem2_b, out + OUT1, (long)HD, mem2t, LM, DD);

    // 4. attention logits + softmax over Lm
    att_kernel<<<dim3(LD / 16, BN), 256, 0, stream>>>(
        inBF, memBF, in_dot, mem_dot, mask, rowmax, Pbuf);

    // 5-6. weight_two softmax + output_two GEMV
    w2_kernel<<<BN, 256, 0, stream>>>(rowmax, w2v);
    otwo_kernel<<<dim3(HD / 256, BN), 256, 0, stream>>>(out, w2v, otwo);

    // 7. output_one GEMM + fused final channels
    out_gemm_kernel<<<dim3(LD / 64, HD / 128, BN), 256, 0, stream>>>(
        Pbuf, mem2t, otwo, out);
}
